// Baseline_v4_74947179315922
// MI455X (gfx1250) — matrix operand order repaired and checked by reading, not compiled, not run on hardware
//
#include <hip/hip_runtime.h>

// MI455X / gfx1250. wave32, WMMA f32_16x16x32_bf16, fp32 accumulate.
// Split-precision bf16 (hi+lo, 3 WMMAs per K-step) ~= fp32 accuracy at ~2.7x
// the FLOP rate of the fp32 16x16x4 WMMA path. Workload is compute-bound
// (~690 GFLOP over ~100 MB => ~7000 FLOP/B), so matrix-pipe throughput rules.

typedef __attribute__((ext_vector_type(16))) __bf16 v16bf;
typedef __attribute__((ext_vector_type(8)))  float  v8f;

struct __attribute__((aligned(16))) U4 { unsigned int x, y, z, w; };

union Frag {
  U4 q[2];
  unsigned int u[8];
  v16bf v;
};

__device__ __forceinline__ unsigned short f2bf(float f) {
  unsigned int u = __float_as_uint(f);
  u += 0x7fffu + ((u >> 16) & 1u);               // round-to-nearest-even
  return (unsigned short)(u >> 16);
}
__device__ __forceinline__ float bf2f(unsigned short h) {
  return __uint_as_float(((unsigned int)h) << 16);
}

// 31 iterations of circular pad-1 on a length-32 axis -> length-94 index map.
// Center: p-31 for p in [31,62]. Borders alternate 31,0,31,... outward.
__device__ __forceinline__ int wrapidx(int p) {
  if (p >= 31 && p <= 62) return p - 31;
  if (p < 31) return ((31 - p) & 1) ? 31 : 0;
  return ((p - 62) & 1) ? 0 : 31;
}

#define ROWS 38   // padded rows staged per ci-block: ry in [2*oy0, 2*oy0+37]
#define RS   96   // LDS row stride (elements)

// dst[b,co,oy,ox] = act( sum_{ci,ky,kx} pad(src)[b,ci,2oy+ky,2ox+kx]*w[co,ci,ky,kx] + bias[co] )
// GEMM view: M=co(32) x N=(b,oy,ox)(16384) x K=(ci,ky,kx)(CI*1024).
// Grid: 128 blocks = (b:16) x (oy-block:8, 4 rows each). Block: 256 thr = 8 waves.
// Wave roles: ks = K-half, mt = co-tile, nh = ox-half; 4 oy accumulators per wave
// (A weight fragment reused across the 4 oy tiles => 4x less L2 weight traffic).
template <int CI>
__global__ __launch_bounds__(256)
void conv_wmma_layer(const float* __restrict__ src,
                     const unsigned short* __restrict__ whi,
                     const unsigned short* __restrict__ wlo,
                     const float* __restrict__ bias,
                     float* __restrict__ dst,
                     int relu)
{
  constexpr int HALF   = (CI == 1) ? 1 : (CI / 2);
  constexpr int NCHUNK = (CI == 1) ? 1 : (CI / 2);
  constexpr int NBLK   = (CI == 1) ? 1 : 2;
  constexpr int NSTEP  = (CI == 1) ? 16 : 32;

  __shared__ unsigned short sHi[2 * ROWS * RS];
  __shared__ unsigned short sLo[2 * ROWS * RS];
  __shared__ float sRed[2][2][4][256];           // [mt][nh][q][lane*8+r] from ks==1

  const int tid  = threadIdx.x;
  const int lane = tid & 31;
  const int wv   = tid >> 5;
  const int ks = wv >> 2;
  const int mt = (wv >> 1) & 1;
  const int nh = wv & 1;
  const int b   = blockIdx.x >> 3;
  const int oy0 = (blockIdx.x & 7) << 2;

  const int l15 = lane & 15;
  const int lh  = lane >> 4;
  const int co  = mt * 16 + l15;

  v8f acc[4];
  #pragma unroll
  for (int q = 0; q < 4; ++q) acc[q] = (v8f){0,0,0,0,0,0,0,0};

  // B fragment (32x16 layout): lane = N(=ox), elements e -> K = 16*lh + e = kx
  const int px0 = 2 * (nh * 16 + l15) + 16 * lh;           // even => 4B aligned

  for (int c = 0; c < NCHUNK; ++c) {
    // ---- stage padded activation rows into LDS as bf16 hi/lo ----
    constexpr int TOTAL = NBLK * ROWS * 94;
    for (int i = tid; i < TOTAL; i += 256) {
      int blk = i / (ROWS * 94);
      int rem = i - blk * (ROWS * 94);
      int row = rem / 94;
      int px  = rem - row * 94;
      int ci  = (CI == 1) ? 0 : (blk * HALF + c);
      int r   = wrapidx(2 * oy0 + row);
      int cc  = wrapidx(px);
      float v = src[((b * CI + ci) * 32 + r) * 32 + cc];
      unsigned short hi = f2bf(v);
      unsigned short lo = f2bf(v - bf2f(hi));
      sHi[blk * (ROWS * RS) + row * RS + px] = hi;
      sLo[blk * (ROWS * RS) + row * RS + px] = lo;
    }
    __syncthreads();

    const int myblk = (CI == 1) ? 0 : ks;
    const int ci    = (CI == 1) ? 0 : (ks * HALF + c);
    const int ky0   = (CI == 1) ? ks * 16 : 0;

    // A fragment (16x32 layout): lane = M, K = 16h + e (contiguous, matches B)
    const int abase0 = ((co * CI + ci) * 32 + ky0) * 32 + 16 * lh;
    const unsigned short* pHi = whi + abase0;
    const unsigned short* pLo = wlo + abase0;
    int rowoff = myblk * (ROWS * RS) + ky0 * RS + px0;

    #pragma unroll 2
    for (int s = 0; s < NSTEP; ++s) {
      Frag ahi, alo;
      ahi.q[0] = *(const U4*)(pHi);
      ahi.q[1] = *(const U4*)(pHi + 8);
      alo.q[0] = *(const U4*)(pLo);
      alo.q[1] = *(const U4*)(pLo + 8);
      pHi += 32; pLo += 32;                                // next ky row of weights

      int ro = rowoff;
      #pragma unroll
      for (int q = 0; q < 4; ++q) {
        Frag bhi, blo;
        const unsigned int* ph = (const unsigned int*)&sHi[ro];
        const unsigned int* pl = (const unsigned int*)&sLo[ro];
        #pragma unroll
        for (int j = 0; j < 8; ++j) { bhi.u[j] = ph[j]; blo.u[j] = pl[j]; }
        ro += 2 * RS;                                      // next oy tile: row = 2q + ky

        acc[q] = __builtin_amdgcn_wmma_f32_16x16x32_bf16(false, ahi.v, false, bhi.v,
                                                         (short)0, acc[q], false, false);
        acc[q] = __builtin_amdgcn_wmma_f32_16x16x32_bf16(false, ahi.v, false, blo.v,
                                                         (short)0, acc[q], false, false);
        acc[q] = __builtin_amdgcn_wmma_f32_16x16x32_bf16(false, alo.v, false, bhi.v,
                                                         (short)0, acc[q], false, false);
      }
      rowoff += RS;                                        // next ky
    }
    __syncthreads();
  }

  // ---- reduce K halves, add bias, relu, store ----
  if (ks == 1) {
    #pragma unroll
    for (int q = 0; q < 4; ++q)
      #pragma unroll
      for (int r = 0; r < 8; ++r)
        sRed[mt][nh][q][lane * 8 + r] = acc[q][r];
  }
  __syncthreads();
  if (ks == 0) {
    #pragma unroll
    for (int q = 0; q < 4; ++q) {
      const int oy = oy0 + q;
      #pragma unroll
      for (int r = 0; r < 8; ++r) {
        const int cc = mt * 16 + 8 * lh + r;               // C/D layout: M = r + 8*(lane>>4)
        float v = acc[q][r] + sRed[mt][nh][q][lane * 8 + r] + bias[cc];
        if (relu) v = fmaxf(v, 0.0f);
        dst[((b * 32 + cc) * 32 + oy) * 32 + (nh * 16 + l15)] = v;
      }
    }
  }
}

// fp32 weights -> bf16 hi/lo pair (in_w then convs_w, packed)
__global__ __launch_bounds__(256)
void cvt_weights(const float* __restrict__ w0, const float* __restrict__ w1,
                 unsigned short* __restrict__ whi, unsigned short* __restrict__ wlo,
                 int n0, int ntot)
{
  for (int i = blockIdx.x * 256 + threadIdx.x; i < ntot; i += gridDim.x * 256) {
    float v = (i < n0) ? w0[i] : w1[i - n0];
    unsigned short hi = f2bf(v);
    unsigned short lo = f2bf(v - bf2f(hi));
    whi[i] = hi; wlo[i] = lo;
  }
}

// 1x1 output conv (tiny; plain VALU is fine)
__global__ __launch_bounds__(256)
void out_conv1x1(const float* __restrict__ h, const float* __restrict__ w,
                 const float* __restrict__ bb, float* __restrict__ out)
{
  const int i = blockIdx.x * 256 + threadIdx.x;   // 16384 outputs
  const int bidx = i >> 10, px = i & 1023;
  float a = bb[0];
  #pragma unroll
  for (int cco = 0; cco < 32; ++cco) a += h[(bidx * 32 + cco) * 1024 + px] * w[cco];
  out[i] = a;
}

extern "C" void kernel_launch(void* const* d_in, const int* in_sizes, int n_in,
                              void* d_out, int out_size, void* d_ws, size_t ws_size,
                              hipStream_t stream)
{
  const float* x       = (const float*)d_in[0];   // [16,1,32,32]
  const float* in_w    = (const float*)d_in[1];   // [32,1,32,32]
  const float* in_b    = (const float*)d_in[2];   // [32]
  const float* convs_w = (const float*)d_in[3];   // [20,32,32,32,32]
  const float* convs_b = (const float*)d_in[4];   // [20,32]
  const float* out_w   = (const float*)d_in[5];   // [1,32,1,1]
  const float* out_b   = (const float*)d_in[6];   // [1]

  const int NW = 32768 + 20 * 1048576;            // 21,004,288 weight elems
  unsigned short* whi = (unsigned short*)d_ws;
  unsigned short* wlo = whi + NW;
  float* actA = (float*)(wlo + NW);               // 16*32*32*32 floats
  float* actB = actA + 16 * 32 * 32 * 32;
  // ws usage: ~88.2 MB total

  cvt_weights<<<2048, 256, 0, stream>>>(in_w, convs_w, whi, wlo, 32768, NW);

  // in_conv: Ci=1, no relu
  conv_wmma_layer<1><<<128, 256, 0, stream>>>(x, whi, wlo, in_b, actA, 0);

  // 20 hidden layers: Ci=32, relu; ping-pong actA/actB (20 is even -> ends in actA)
  float* cur = actA;
  float* nxt = actB;
  for (int l = 0; l < 20; ++l) {
    const unsigned short* lw_hi = whi + 32768 + (size_t)l * 1048576;
    const unsigned short* lw_lo = wlo + 32768 + (size_t)l * 1048576;
    conv_wmma_layer<32><<<128, 256, 0, stream>>>(cur, lw_hi, lw_lo, convs_b + 32 * l, nxt, 1);
    float* t = cur; cur = nxt; nxt = t;
  }

  out_conv1x1<<<64, 256, 0, stream>>>(cur, out_w, out_b, (float*)d_out);
}
